// NewRec_33595234189646
// MI455X (gfx1250) — matrix operand order repaired and checked by reading, not compiled, not run on hardware
//
#include <hip/hip_runtime.h>
#include <hip/hip_bf16.h>
#include <cstdint>
#include <cstddef>

typedef __attribute__((ext_vector_type(16))) _Float16 v16h;
typedef __attribute__((ext_vector_type(8)))  _Float16 v8h;
typedef __attribute__((ext_vector_type(8)))  float    v8f;
typedef __attribute__((ext_vector_type(4)))  unsigned int v4u;
typedef __attribute__((ext_vector_type(4)))  int v4i;

#define Bb   32
#define Ls   512
#define HIDs 512
#define Hh   8
#define DHs  64
#define NBs  2
#define ITEMP1 20001
#define NEGC (-4294967295.0f)
#define WPAD 40   // LDS row pitch (halves) for 32-wide K tiles: 80B -> conflict-free strides

#define AS1 __attribute__((address_space(1)))
#define AS3 __attribute__((address_space(3)))

#if defined(__has_builtin)
#  if __has_builtin(__builtin_amdgcn_global_load_async_to_lds_b128)
#    define HAVE_ASYNC 1
#  else
#    define HAVE_ASYNC 0
#  endif
#  if __has_builtin(__builtin_amdgcn_s_wait_asynccnt)
#    define HAVE_WAITASYNC 1
#  else
#    define HAVE_WAITASYNC 0
#  endif
#else
#  define HAVE_ASYNC 0
#  define HAVE_WAITASYNC 0
#endif

// ---------------- fragment loaders (CDNA5 16-bit WMMA layouts) ----------------

// A-matrix 16x32 (MxK): lanes 0-15 -> M, K 0..15; lanes 16-31 -> K 16..31
__device__ __forceinline__ v16h load_a_frag(const _Float16* A, int lda, int m0, int k0, int lane) {
  const int m  = m0 + (lane & 15);
  const int kb = k0 + ((lane >> 4) << 4);
  const _Float16* p = A + (size_t)m * lda + kb;
  v16h r;
#pragma unroll
  for (int i = 0; i < 8; ++i) r[i] = p[i];
#pragma unroll
  for (int i = 0; i < 8; ++i) r[8 + i] = p[8 + i];
  return r;
}

// B-matrix 32x16 (KxN) from W stored (N x K) row-major: 16 contiguous halves per lane.
__device__ __forceinline__ v16h load_bT_frag(const _Float16* W, int ldw, int n0, int k0, int lane) {
  const int n  = n0 + (lane & 15);
  const int kb = k0 + ((lane >> 4) << 4);
  const _Float16* p = W + (size_t)n * ldw + kb;
  v16h r;
#pragma unroll
  for (int i = 0; i < 16; ++i) r[i] = p[i];
  return r;
}

// Same, but from an LDS-staged 64 x 32 tile with WPAD row pitch.
__device__ __forceinline__ v16h load_b_lds(const _Float16* Wl, int n0, int lane) {
  const int n  = n0 + (lane & 15);
  const int kb = (lane >> 4) << 4;
  const _Float16* p = Wl + n * WPAD + kb;
  v16h r;
#pragma unroll
  for (int i = 0; i < 16; ++i) r[i] = p[i];
  return r;
}

// B-matrix 32x16 (KxN) from a row-major K x N matrix (e.g. V): strided gather down a column.
__device__ __forceinline__ v16h load_b_frag(const _Float16* Bm, int ldb, int k0, int n0, int lane) {
  const int n  = n0 + (lane & 15);
  const int kb = k0 + ((lane >> 4) << 4);
  v16h r;
#pragma unroll
  for (int i = 0; i < 16; ++i) r[i] = Bm[(size_t)(kb + i) * ldb + n];
  return r;
}

__device__ __forceinline__ float wave_reduce_add(float v) {
#pragma unroll
  for (int off = 16; off; off >>= 1) v += __shfl_xor(v, off, 32);
  return v;
}
__device__ __forceinline__ float wave_reduce_max(float v) {
#pragma unroll
  for (int off = 16; off; off >>= 1) v = fmaxf(v, __shfl_xor(v, off, 32));
  return v;
}

// Stage a 64(N) x 32(K) W tile into LDS; 256 16-byte chunks over 128 threads.
__device__ __forceinline__ void stage_w(const _Float16* W, int K, int n0, int k0,
                                        _Float16* dst, int tid) {
#pragma unroll
  for (int c = 0; c < 2; ++c) {
    const int chunk = tid + c * 128;        // 0..255
    const int n  = chunk >> 2;              // 0..63
    const int ko = (chunk & 3) << 3;        // 0,8,16,24 halves
    const _Float16* src = W + (size_t)(n0 + n) * K + k0 + ko;
    _Float16* d = dst + n * WPAD + ko;
#if HAVE_ASYNC
    __builtin_amdgcn_global_load_async_to_lds_b128((AS1 v4i*)src, (AS3 v4i*)d, 0, 0);
#else
    *(v4u*)d = *(const v4u*)src;
#endif
  }
}

__device__ __forceinline__ void async_wait_barrier() {
#if HAVE_ASYNC
#  if HAVE_WAITASYNC
  __builtin_amdgcn_s_wait_asynccnt(0);
#  else
  asm volatile("s_wait_asynccnt 0x0" ::: "memory");
#  endif
#endif
  __syncthreads();
}

// ---------------- generic GEMM: C = act(A @ W^T + bias [+ resid]) ----------------
// block = 128 threads (4 waves); block tile 128(M) x 64(N); wave tile 32 x 64.
// W tiles double-buffered in LDS via async loads; A streamed direct + prefetch.
__global__ __launch_bounds__(128) void gemm_wmma_kernel(
    const _Float16* __restrict__ A, const _Float16* __restrict__ W,
    const float* __restrict__ bias, const float* __restrict__ resid,
    float* __restrict__ Cf, _Float16* __restrict__ Ch,
    int M, int N, int K, int relu)
{
  __shared__ _Float16 wt[2][64 * WPAD];
  const int tid  = threadIdx.x;
  const int lane = tid & 31;
  const int wave = tid >> 5;
  const int n0 = blockIdx.x * 64;
  const int m0 = blockIdx.y * 128 + wave * 32;

  v8f acc[2][4] = {};
  const int nk = K >> 5;
  stage_w(W, K, n0, 0, &wt[0][0], tid);
  for (int kt = 0; kt < nk; ++kt) {
    async_wait_barrier();
    if (kt + 1 < nk) stage_w(W, K, n0, (kt + 1) << 5, &wt[(kt + 1) & 1][0], tid);
    if (kt + 2 < nk)
      __builtin_prefetch(A + (size_t)(m0 + (lane & 15)) * K + ((kt + 2) << 5), 0, 0);
    v16h a0 = load_a_frag(A, K, m0,      kt << 5, lane);
    v16h a1 = load_a_frag(A, K, m0 + 16, kt << 5, lane);
    const _Float16* wl = &wt[kt & 1][0];
#pragma unroll
    for (int t = 0; t < 4; ++t) {
      v16h b = load_b_lds(wl, 16 * t, lane);
      acc[0][t] = __builtin_amdgcn_wmma_f32_16x16x32_f16(false, a0, false, b, (short)0, acc[0][t], false, false);
      acc[1][t] = __builtin_amdgcn_wmma_f32_16x16x32_f16(false, a1, false, b, (short)0, acc[1][t], false, false);
    }
  }
  const int hi = (lane >> 4) << 3;
  const int nc = lane & 15;
#pragma unroll
  for (int half = 0; half < 2; ++half) {
#pragma unroll
    for (int t = 0; t < 4; ++t) {
      const int col = n0 + 16 * t + nc;
      const float bv = bias ? bias[col] : 0.0f;
#pragma unroll
      for (int r = 0; r < 8; ++r) {
        const size_t row = (size_t)(m0 + 16 * half + hi + r);
        float v = acc[half][t][r] + bv;
        if (resid) v += resid[row * N + col];
        if (relu)  v = fmaxf(v, 0.0f);
        if (Cf) Cf[row * N + col] = v;
        if (Ch) Ch[row * N + col] = (_Float16)v;
      }
    }
  }
}

// ---------------- fused attention: one wave per 16-query strip per (b,h) ----------------
__global__ __launch_bounds__(32) void attn_kernel(
    const _Float16* __restrict__ q, const _Float16* __restrict__ k,
    const _Float16* __restrict__ v, float* __restrict__ out)
{
  __shared__ float    sS[16 * Ls];
  __shared__ _Float16 sP[16 * Ls];
  const int lane = threadIdx.x;
  const int id = blockIdx.x;
  const int lt = id & 31;
  const int h  = (id >> 5) & 7;
  const int b  = id >> 8;
  const int l0 = lt * 16;
  const int mE = (l0 + 16 + 31) & ~31;   // causal bound, rounded to K-step
  const size_t base = ((size_t)b * Ls) * HIDs + h * DHs;
  const _Float16* Qp = q + base;
  const _Float16* Kp = k + base;
  const _Float16* Vp = v + base;

  // Phase 1: S = Q K^T / sqrt(DH), causal mask, into LDS
  for (int m0 = 0; m0 < mE; m0 += 16) {
    v8f acc = {};
#pragma unroll
    for (int k0 = 0; k0 < DHs; k0 += 32) {
      v16h a  = load_a_frag(Qp, HIDs, l0, k0, lane);
      v16h bb = load_bT_frag(Kp, HIDs, m0, k0, lane);
      acc = __builtin_amdgcn_wmma_f32_16x16x32_f16(false, a, false, bb, (short)0, acc, false, false);
    }
    const int hi = (lane >> 4) << 3;
    const int mc = m0 + (lane & 15);
#pragma unroll
    for (int r = 0; r < 8; ++r) {
      const int lr = l0 + hi + r;
      float s = acc[r] * 0.125f;
      if (mc > lr) s = NEGC;
      sS[(hi + r) * Ls + mc] = s;
    }
  }
  __syncthreads();

  // Phase 2: row softmax over valid prefix, write P (f16) to LDS
  for (int r = 0; r < 16; ++r) {
    float mx = -3.0e38f;
    for (int c = lane; c < mE; c += 32) mx = fmaxf(mx, sS[r * Ls + c]);
    mx = wave_reduce_max(mx);
    float sum = 0.0f;
    for (int c = lane; c < mE; c += 32) {
      float e = __expf(sS[r * Ls + c] - mx);
      sS[r * Ls + c] = e;
      sum += e;
    }
    sum = wave_reduce_add(sum);
    const float inv = 1.0f / sum;
    for (int c = lane; c < mE; c += 32) sP[r * Ls + c] = (_Float16)(sS[r * Ls + c] * inv);
  }
  __syncthreads();

  // Phase 3: out = P @ V over valid prefix
  v8f oacc[4] = {};
  for (int k0 = 0; k0 < mE; k0 += 32) {
    v16h a = load_a_frag((const _Float16*)sP, Ls, 0, k0, lane);
#pragma unroll
    for (int t = 0; t < 4; ++t) {
      v16h bb = load_b_frag(Vp, HIDs, k0, t * 16, lane);
      oacc[t] = __builtin_amdgcn_wmma_f32_16x16x32_f16(false, a, false, bb, (short)0, oacc[t], false, false);
    }
  }
  const int hi = (lane >> 4) << 3;
  const int nc = lane & 15;
#pragma unroll
  for (int t = 0; t < 4; ++t)
#pragma unroll
    for (int r = 0; r < 8; ++r)
      out[((size_t)b * Ls + l0 + hi + r) * HIDs + h * DHs + t * 16 + nc] = oacc[t][r];
}

// ---------------- LayerNorm (one wave per 512-wide row) ----------------
__global__ __launch_bounds__(256) void ln_kernel(
    const float* __restrict__ x, const float* __restrict__ g, const float* __restrict__ bta,
    float* __restrict__ yf, _Float16* __restrict__ yh, int rows)
{
  const int lane = threadIdx.x & 31;
  const int wv = threadIdx.x >> 5;
  const int row = blockIdx.x * 8 + wv;
  if (row >= rows) return;
  const float* xr = x + (size_t)row * HIDs;
  float s = 0.f;
  for (int c = lane; c < HIDs; c += 32) s += xr[c];
  s = wave_reduce_add(s);
  const float mean = s * (1.0f / HIDs);
  float vv = 0.f;
  for (int c = lane; c < HIDs; c += 32) { float d = xr[c] - mean; vv += d * d; }
  vv = wave_reduce_add(vv);
  const float rstd = rsqrtf(vv * (1.0f / HIDs) + 1e-8f);
  for (int c = lane; c < HIDs; c += 32) {
    float o = g[c] * (xr[c] - mean) * rstd + bta[c];
    yf[(size_t)row * HIDs + c] = o;
    if (yh) yh[(size_t)row * HIDs + c] = (_Float16)o;
  }
}

// ---------------- elementwise helpers ----------------
__global__ void f32_to_f16_kernel(const float* __restrict__ in, _Float16* __restrict__ out, size_t n) {
  size_t i = (size_t)blockIdx.x * blockDim.x + threadIdx.x;
  if (i < n) out[i] = (_Float16)in[i];
}

__global__ void pop_enc_kernel(const int* __restrict__ seqs, const int* __restrict__ t1,
                               const int* __restrict__ t2, const float* __restrict__ mt,
                               const float* __restrict__ wt, _Float16* __restrict__ X) {
  size_t i = (size_t)blockIdx.x * blockDim.x + threadIdx.x; // over B*L*128
  if (i >= (size_t)Bb * Ls * 128) return;
  const int j = (int)(i & 127);
  const size_t tok = i >> 7;
  const int sv = seqs[tok];
  float v;
  if (j < 64) v = mt[((size_t)(t1[tok] * 64 + j)) * ITEMP1 + sv];
  else        v = wt[((size_t)(t2[tok] * 64 + (j - 64))) * ITEMP1 + sv];
  X[i] = (_Float16)v;
}

__global__ void addpos_mask_kernel(const float* __restrict__ E, const float* __restrict__ post,
                                   const int* __restrict__ logs, float* __restrict__ sf,
                                   _Float16* __restrict__ sh) {
  size_t i = (size_t)blockIdx.x * blockDim.x + threadIdx.x; // over B*L*HID
  if (i >= (size_t)Bb * Ls * HIDs) return;
  const int c = (int)(i % HIDs);
  const size_t row = i / HIDs;
  const int l = (int)(row % Ls);
  float v = E[i] + post[(size_t)l * HIDs + c];
  if (logs[row] == 0) v = 0.0f;
  sf[i] = v;
  sh[i] = (_Float16)v;
}

__global__ void add_kernel(const float* __restrict__ a, const float* __restrict__ b,
                           float* __restrict__ o, size_t n) {
  size_t i = (size_t)blockIdx.x * blockDim.x + threadIdx.x;
  if (i < n) o[i] = a[i] + b[i];
}

__global__ void mask_kernel(const float* __restrict__ x, const int* __restrict__ logs,
                            float* __restrict__ sf, _Float16* __restrict__ sh) {
  size_t i = (size_t)blockIdx.x * blockDim.x + threadIdx.x;
  if (i >= (size_t)Bb * Ls * HIDs) return;
  const size_t row = i / HIDs;
  float v = x[i];
  if (logs[row] == 0) v = 0.0f;
  sf[i] = v;
  sh[i] = (_Float16)v;
}

__global__ void finalize_feats_kernel(const float* __restrict__ feats, const int* __restrict__ pu,
                                      const int* __restrict__ nu, float* __restrict__ out_last,
                                      float* __restrict__ out_pos, float* __restrict__ out_neg) {
  int i = blockIdx.x * blockDim.x + threadIdx.x; // B*HID
  if (i >= Bb * HIDs) return;
  const int c = i % HIDs;
  const int b = i / HIDs;
  out_last[i] = feats[((size_t)b * Ls + Ls - 1) * HIDs + c];
  out_pos[i]  = feats[((size_t)pu[b] * Ls + Ls - 1) * HIDs + c];
  out_neg[i]  = feats[((size_t)nu[b] * Ls + Ls - 1) * HIDs + c];
}

__global__ __launch_bounds__(256) void dot_kernel(const float* __restrict__ a,
                                                  const float* __restrict__ b,
                                                  float* __restrict__ o, int rows) {
  const int lane = threadIdx.x & 31;
  const int wv = threadIdx.x >> 5;
  const int row = blockIdx.x * 8 + wv;
  if (row >= rows) return;
  const float* ar = a + (size_t)row * HIDs;
  const float* br = b + (size_t)row * HIDs;
  float s = 0.f;
  for (int c = lane; c < HIDs; c += 32) s += ar[c] * br[c];
  s = wave_reduce_add(s);
  if (lane == 0) o[row] = s;
}

// ---------------- host driver ----------------
extern "C" void kernel_launch(void* const* d_in, const int* in_sizes, int n_in,
                              void* d_out, int out_size, void* d_ws, size_t ws_size,
                              hipStream_t stream) {
  const int*   log_seqs  = (const int*)d_in[0];
  const int*   t1        = (const int*)d_in[1];
  const int*   t2        = (const int*)d_in[2];
  const int*   pos_seqs  = (const int*)d_in[3];
  const int*   neg_seqs  = (const int*)d_in[4];
  const int*   pos_user  = (const int*)d_in[5];
  const int*   neg_user  = (const int*)d_in[6];
  const float* month_t   = (const float*)d_in[7];
  const float* week_t    = (const float*)d_in[8];
  const float* pos_table = (const float*)d_in[9];
  const float* fc1_w     = (const float*)d_in[10];
  const float* fc1_b     = (const float*)d_in[11];
  const float* fc2_w     = (const float*)d_in[12];
  const float* fc2_b     = (const float*)d_in[13];
  const float* qw        = (const float*)d_in[14];
  const float* qb        = (const float*)d_in[15];
  const float* kw        = (const float*)d_in[16];
  const float* kb        = (const float*)d_in[17];
  const float* vw        = (const float*)d_in[18];
  const float* vb        = (const float*)d_in[19];
  const float* aln_g     = (const float*)d_in[20];
  const float* aln_b     = (const float*)d_in[21];
  const float* fln_g     = (const float*)d_in[22];
  const float* fln_b     = (const float*)d_in[23];
  const float* c1w       = (const float*)d_in[24];
  const float* c1b       = (const float*)d_in[25];
  const float* c2w       = (const float*)d_in[26];
  const float* c2b       = (const float*)d_in[27];
  const float* lln_g     = (const float*)d_in[28];
  const float* lln_b     = (const float*)d_in[29];
  float* out = (float*)d_out;

  const size_t BL = (size_t)Bb * Ls;
  char* ws = (char*)d_ws;
  size_t off = 0;
  auto alloc = [&](size_t bytes) -> void* {
    off = (off + 255) & ~(size_t)255;
    void* p = ws + off;
    off += bytes;
    return p;
  };

  _Float16* fc1h = (_Float16*)alloc(1024 * 128 * 2);
  _Float16* fc2h = (_Float16*)alloc(512 * 1024 * 2);
  _Float16* qwh  = (_Float16*)alloc((size_t)NBs * 512 * 512 * 2);
  _Float16* kwh  = (_Float16*)alloc((size_t)NBs * 512 * 512 * 2);
  _Float16* vwh  = (_Float16*)alloc((size_t)NBs * 512 * 512 * 2);
  _Float16* c1wh = (_Float16*)alloc((size_t)NBs * 512 * 512 * 2);
  _Float16* c2wh = (_Float16*)alloc((size_t)NBs * 512 * 512 * 2);
  _Float16* Xh   = (_Float16*)alloc(BL * 128 * 2);
  void*     big  = alloc(BL * 1024 * 2 + BL * HIDs * 2);   // H1 f16 (embed) aliased with tmp f32 (blocks)
  _Float16* H1h  = (_Float16*)big;
  float*    tmpf = (float*)big;
  float*    Ef   = (float*)alloc(BL * HIDs * 4);
  float*    sf   = (float*)alloc(BL * HIDs * 4);
  _Float16* sh   = (_Float16*)alloc(BL * HIDs * 2);
  float*    Qf   = (float*)alloc(BL * HIDs * 4);            // also final log_feats
  _Float16* Qh   = (_Float16*)alloc(BL * HIDs * 2);
  _Float16* qh   = (_Float16*)alloc(BL * HIDs * 2);         // also FFN hidden f16
  _Float16* kh   = (_Float16*)alloc(BL * HIDs * 2);
  _Float16* vh   = (_Float16*)alloc(BL * HIDs * 2);

  auto cvt = [&](const float* in, _Float16* o, size_t n) {
    f32_to_f16_kernel<<<dim3((unsigned)((n + 255) / 256)), dim3(256), 0, stream>>>(in, o, n);
  };
  auto gemm = [&](const _Float16* A, const _Float16* W, const float* bias, const float* resid,
                  float* Cf, _Float16* Ch, int M, int N, int K, int relu) {
    gemm_wmma_kernel<<<dim3(N / 64, M / 128), dim3(128), 0, stream>>>(A, W, bias, resid, Cf, Ch, M, N, K, relu);
  };
  auto ln = [&](const float* x, const float* g, const float* b, float* yf, _Float16* yh) {
    ln_kernel<<<dim3((unsigned)(BL / 8)), dim3(256), 0, stream>>>(x, g, b, yf, yh, (int)BL);
  };

  // --- weight conversion to f16 ---
  cvt(fc1_w, fc1h, 1024 * 128);
  cvt(fc2_w, fc2h, 512 * 1024);
  cvt(qw,  qwh,  (size_t)NBs * 512 * 512);
  cvt(kw,  kwh,  (size_t)NBs * 512 * 512);
  cvt(vw,  vwh,  (size_t)NBs * 512 * 512);
  cvt(c1w, c1wh, (size_t)NBs * 512 * 512);
  cvt(c2w, c2wh, (size_t)NBs * 512 * 512);

  const unsigned ew_grid = (unsigned)((BL * HIDs + 255) / 256);
  const unsigned pe_grid = (unsigned)((BL * 128 + 255) / 256);

  // --- seqs = embed(pop_enc(log_seqs)) + pos_table, masked ---
  pop_enc_kernel<<<pe_grid, 256, 0, stream>>>(log_seqs, t1, t2, month_t, week_t, Xh);
  gemm(Xh, fc1h, fc1_b, nullptr, nullptr, H1h, (int)BL, 1024, 128, 1);
  gemm(H1h, fc2h, fc2_b, nullptr, Ef, nullptr, (int)BL, 512, 1024, 0);
  addpos_mask_kernel<<<ew_grid, 256, 0, stream>>>(Ef, pos_table, log_seqs, sf, sh);

  // --- transformer blocks ---
  for (int i = 0; i < NBs; ++i) {
    const size_t wo = (size_t)i * 512 * 512;
    const size_t bo = (size_t)i * 512;
    ln(sf, aln_g + bo, aln_b + bo, Qf, Qh);
    gemm(Qh, qwh + wo, qb + bo, nullptr, nullptr, qh, (int)BL, 512, 512, 0);
    gemm(sh, kwh + wo, kb + bo, nullptr, nullptr, kh, (int)BL, 512, 512, 0);
    gemm(sh, vwh + wo, vb + bo, nullptr, nullptr, vh, (int)BL, 512, 512, 0);
    attn_kernel<<<dim3(Bb * Hh * (Ls / 16)), dim3(32), 0, stream>>>(qh, kh, vh, tmpf);
    add_kernel<<<ew_grid, 256, 0, stream>>>(Qf, tmpf, sf, BL * HIDs);
    ln(sf, fln_g + bo, fln_b + bo, sf, sh);
    gemm(sh, c1wh + wo, c1b + bo, nullptr, nullptr, qh, (int)BL, 512, 512, 1);   // FFN hidden (f16)
    gemm(qh, c2wh + wo, c2b + bo, sf, tmpf, nullptr, (int)BL, 512, 512, 0);      // + residual
    mask_kernel<<<ew_grid, 256, 0, stream>>>(tmpf, log_seqs, sf, sh);
  }

  // --- final LN -> log_feats (Qf) ---
  ln(sf, lln_g, lln_b, Qf, nullptr);
  finalize_feats_kernel<<<dim3((Bb * HIDs + 255) / 256), dim3(256), 0, stream>>>(
      Qf, pos_user, neg_user, out + 2 * 16384, out + 3 * 16384, out + 4 * 16384);

  // --- pos logits ---
  pop_enc_kernel<<<pe_grid, 256, 0, stream>>>(pos_seqs, t1, t2, month_t, week_t, Xh);
  gemm(Xh, fc1h, fc1_b, nullptr, nullptr, H1h, (int)BL, 1024, 128, 1);
  gemm(H1h, fc2h, fc2_b, nullptr, Ef, nullptr, (int)BL, 512, 1024, 0);
  dot_kernel<<<dim3((unsigned)(BL / 8)), dim3(256), 0, stream>>>(Qf, Ef, out, (int)BL);

  // --- neg logits ---
  pop_enc_kernel<<<pe_grid, 256, 0, stream>>>(neg_seqs, t1, t2, month_t, week_t, Xh);
  gemm(Xh, fc1h, fc1_b, nullptr, nullptr, H1h, (int)BL, 1024, 128, 1);
  gemm(H1h, fc2h, fc2_b, nullptr, Ef, nullptr, (int)BL, 512, 1024, 0);
  dot_kernel<<<dim3((unsigned)(BL / 8)), dim3(256), 0, stream>>>(Qf, Ef, out + 16384, (int)BL);
}
